// Attention1x1_54743653155442
// MI455X (gfx1250) — matrix operand order repaired and checked by reading, not compiled, not run on hardware
//
#include <hip/hip_runtime.h>

// ---------------------------------------------------------------------------
// Types for CDNA5 WMMA (wave32): A/B = 16 bf16 per lane, C/D = 8 f32 per lane
// ---------------------------------------------------------------------------
typedef __attribute__((ext_vector_type(16))) __bf16        v16bf;
typedef __attribute__((ext_vector_type(16))) unsigned short v16us;
typedef __attribute__((ext_vector_type(8)))  float          v8f;

__device__ __forceinline__ unsigned short f2bf(float f) {
  unsigned int u = __builtin_bit_cast(unsigned int, f);
  u += 0x7FFFu + ((u >> 16) & 1u);        // round-to-nearest-even
  return (unsigned short)(u >> 16);
}
__device__ __forceinline__ float bf2f(unsigned short u) {
  return __builtin_bit_cast(float, (unsigned int)u << 16);
}

__device__ __forceinline__ void put8(v16us& f, int base, uint4 d) {
  f[base + 0] = (unsigned short)d.x; f[base + 1] = (unsigned short)(d.x >> 16);
  f[base + 2] = (unsigned short)d.y; f[base + 3] = (unsigned short)(d.y >> 16);
  f[base + 4] = (unsigned short)d.z; f[base + 5] = (unsigned short)(d.z >> 16);
  f[base + 6] = (unsigned short)d.w; f[base + 7] = (unsigned short)(d.w >> 16);
}

__device__ __forceinline__ v8f wmma_bf16(v16us a, v16us b, v8f c) {
  return __builtin_amdgcn_wmma_f32_16x16x32_bf16(
      /*neg_a=*/false, __builtin_bit_cast(v16bf, a),
      /*neg_b=*/false, __builtin_bit_cast(v16bf, b),
      /*c_mod=*/(short)0, c, /*reuse_a=*/false, /*reuse_b=*/false);
}

// CDNA5 async copy: global -> LDS, 16 bytes per lane, tracked by ASYNCcnt.
__device__ __forceinline__ void async_copy_b128(void* lds_dst, const void* gsrc) {
  unsigned lds_addr = (unsigned)(size_t)lds_dst;                 // LDS aperture: offset in ADDR[31:0]
  unsigned long long ga = (unsigned long long)(size_t)gsrc;      // 64-bit vaddr pair
  asm volatile("global_load_async_to_lds_b128 %0, %1, off"
               :: "v"(lds_addr), "v"(ga) : "memory");
}
__device__ __forceinline__ void wait_async0() {
  asm volatile("s_wait_asynccnt 0" ::: "memory");
}

// ---------------------------------------------------------------------------
// f32 -> bf16 bulk convert (8 elems/thread, b128 in / b128 out)
// ---------------------------------------------------------------------------
__global__ __launch_bounds__(256) void k_f32_to_bf16(
    const float* __restrict__ in, unsigned short* __restrict__ out, long long n8) {
  long long idx = (long long)blockIdx.x * 256 + threadIdx.x;
  if (idx >= n8) return;
  const float4* p = (const float4*)in + idx * 2;
  float4 a = p[0], b = p[1];
  uint4 o;
  o.x = (unsigned)f2bf(a.x) | ((unsigned)f2bf(a.y) << 16);
  o.y = (unsigned)f2bf(a.z) | ((unsigned)f2bf(a.w) << 16);
  o.z = (unsigned)f2bf(b.x) | ((unsigned)f2bf(b.y) << 16);
  o.w = (unsigned)f2bf(b.z) | ((unsigned)f2bf(b.w) << 16);
  ((uint4*)out)[idx] = o;
}

// ---------------------------------------------------------------------------
// Generic bf16 GEMM, B row-major [K x N]. Block = 8 waves, block tile 128Mx64N,
// wave tile 16Mx64N (one A frag reused across 4 WMMA).
// B panel is double-buffered in LDS via async global->LDS copies (ASYNCcnt):
// tile i+1 streams in while the 4 WMMAs consume tile i.
// ---------------------------------------------------------------------------
template <bool DW, bool OUT_BF16>
__global__ __launch_bounds__(256) void gemm_rowB(
    const unsigned short* __restrict__ A, long long strideA,
    const unsigned short* __restrict__ Bm, long long strideB,
    const float* __restrict__ dw,
    void* __restrict__ Dp, long long strideD,
    int lda, int ldb, int ldd, int K) {
  const int b    = blockIdx.z;
  const unsigned short* Ab = A  + (long long)b * strideA;
  const unsigned short* Bb = Bm + (long long)b * strideB;
  const int wave = threadIdx.x >> 5;
  const int lane = threadIdx.x & 31;
  const int lh   = lane >> 4;        // 0/1 : lane half
  const int lc   = lane & 15;
  const int mBase = blockIdx.y * 128 + wave * 16;
  const int nBase = blockIdx.x * 64;

  __shared__ unsigned short Bs[2][32][72];  // double-buffered 32K x 64N bf16 tile

  v8f acc0 = {}, acc1 = {}, acc2 = {}, acc3 = {};
  const int ldr = threadIdx.x >> 3;         // 0..31 (k row)
  const int ldc = (threadIdx.x & 7) * 8;    // 0..56 (n chunk), 16B granules

  // prologue: kick off tile 0
  async_copy_b128(&Bs[0][ldr][ldc],
                  Bb + (long long)ldr * ldb + (nBase + ldc));

  const int nIter = K / 32;
  for (int it = 0; it < nIter; ++it) {
    const int k0 = it * 32;
    wait_async0();        // this wave's tile `it` fully in LDS (asyncs in-order)
    __syncthreads();      // all waves done: tile `it` complete, prev buffer free

    if (it + 1 < nIter)   // overlap: stream tile it+1 while computing tile it
      async_copy_b128(&Bs[(it + 1) & 1][ldr][ldc],
                      Bb + (long long)(k0 + 32 + ldr) * ldb + (nBase + ldc));

    // A fragment: row m = mBase+lc; element i <-> k = lh*16 + i
    const unsigned short* arow = Ab + (long long)(mBase + lc) * lda + k0;
    v16us af;
    put8(af, 0, *(const uint4*)(arow + lh * 16));
    put8(af, 8, *(const uint4*)(arow + lh * 16 + 8));

    // B fragments for 4 n-tiles: element i <-> k = lh*16 + i, col = t*16 + lc
    const unsigned short (*Bt)[72] = Bs[it & 1];
    v16us bf0, bf1, bf2, bf3;
#pragma unroll
    for (int i = 0; i < 16; ++i) {
      const unsigned short* brow = &Bt[lh * 16 + i][0];
      bf0[i] = brow[lc];
      bf1[i] = brow[16 + lc];
      bf2[i] = brow[32 + lc];
      bf3[i] = brow[48 + lc];
    }
    acc0 = wmma_bf16(af, bf0, acc0);
    acc1 = wmma_bf16(af, bf1, acc1);
    acc2 = wmma_bf16(af, bf2, acc2);
    acc3 = wmma_bf16(af, bf3, acc3);
  }

  float rs[8];
#pragma unroll
  for (int i = 0; i < 8; ++i) rs[i] = DW ? dw[mBase + lh * 8 + i] : 1.0f;

#pragma unroll
  for (int i = 0; i < 8; ++i) {
    const long long row  = mBase + lh * 8 + i;
    const long long base = (long long)b * strideD + row * ldd + nBase + lc;
    float v0 = acc0[i] * rs[i], v1 = acc1[i] * rs[i];
    float v2 = acc2[i] * rs[i], v3 = acc3[i] * rs[i];
    if (OUT_BF16) {
      unsigned short* o = (unsigned short*)Dp;
      o[base] = f2bf(v0); o[base + 16] = f2bf(v1);
      o[base + 32] = f2bf(v2); o[base + 48] = f2bf(v3);
    } else {
      float* o = (float*)Dp;
      o[base] = v0; o[base + 16] = v1; o[base + 32] = v2; o[base + 48] = v3;
    }
  }
}

// ---------------------------------------------------------------------------
// attn_raw[b][c][d] = sum_n q[c][n]*k[d][n]  (B operand = k^T: per-lane
// contiguous b128 loads along n, no LDS staging needed). K = N = 16384.
// ---------------------------------------------------------------------------
__global__ __launch_bounds__(256) void attn_qk(
    const unsigned short* __restrict__ qkv, float* __restrict__ attn_raw, int N) {
  const int b = blockIdx.z;
  const unsigned short* qb = qkv + (long long)b * 768 * N;
  const unsigned short* kb = qb + 256LL * N;
  const int wave = threadIdx.x >> 5;
  const int lane = threadIdx.x & 31;
  const int lh = lane >> 4, lc = lane & 15;
  const int mBase = blockIdx.y * 128 + wave * 16;
  const int dBase = blockIdx.x * 64;

  v8f acc0 = {}, acc1 = {}, acc2 = {}, acc3 = {};
  const unsigned short* arow  = qb + (long long)(mBase + lc) * N;
  const unsigned short* brow0 = kb + (long long)(dBase + lc) * N;
  const unsigned short* brow1 = brow0 + 16LL * N;
  const unsigned short* brow2 = brow0 + 32LL * N;
  const unsigned short* brow3 = brow0 + 48LL * N;

  for (int k0 = 0; k0 < N; k0 += 32) {
    v16us af, b0, b1, b2, b3;
    put8(af, 0, *(const uint4*)(arow + k0 + lh * 16));
    put8(af, 8, *(const uint4*)(arow + k0 + lh * 16 + 8));
    const int no = k0 + lh * 16;
    put8(b0, 0, *(const uint4*)(brow0 + no)); put8(b0, 8, *(const uint4*)(brow0 + no + 8));
    put8(b1, 0, *(const uint4*)(brow1 + no)); put8(b1, 8, *(const uint4*)(brow1 + no + 8));
    put8(b2, 0, *(const uint4*)(brow2 + no)); put8(b2, 8, *(const uint4*)(brow2 + no + 8));
    put8(b3, 0, *(const uint4*)(brow3 + no)); put8(b3, 8, *(const uint4*)(brow3 + no + 8));
    acc0 = wmma_bf16(af, b0, acc0);
    acc1 = wmma_bf16(af, b1, acc1);
    acc2 = wmma_bf16(af, b2, acc2);
    acc3 = wmma_bf16(af, b3, acc3);
  }
#pragma unroll
  for (int i = 0; i < 8; ++i) {
    long long row  = mBase + lh * 8 + i;
    long long base = ((long long)b * 256 + row) * 256 + dBase + lc;
    attn_raw[base] = acc0[i]; attn_raw[base + 16] = acc1[i];
    attn_raw[base + 32] = acc2[i]; attn_raw[base + 48] = acc3[i];
  }
}

// ---------------------------------------------------------------------------
// Row sum-of-squares for q and k (rows o in [0,512) of qkv) -> sumsq[b*512+o]
// ---------------------------------------------------------------------------
__global__ __launch_bounds__(256) void sumsq_rows(
    const unsigned short* __restrict__ qkv, float* __restrict__ out, int N) {
  __shared__ float red[256];
  int row = blockIdx.x, b = row >> 9, o = row & 511;
  const uint4* p = (const uint4*)(qkv + ((long long)b * 768 + o) * N);
  float s = 0.f;
  for (int c = threadIdx.x; c < N / 8; c += 256) {
    uint4 d = p[c];
    unsigned u[4] = {d.x, d.y, d.z, d.w};
#pragma unroll
    for (int j = 0; j < 4; ++j) {
      float f0 = bf2f((unsigned short)u[j]);
      float f1 = bf2f((unsigned short)(u[j] >> 16));
      s += f0 * f0 + f1 * f1;
    }
  }
  red[threadIdx.x] = s; __syncthreads();
  for (int w = 128; w > 0; w >>= 1) {
    if ((int)threadIdx.x < w) red[threadIdx.x] += red[threadIdx.x + w];
    __syncthreads();
  }
  if (threadIdx.x == 0) out[row] = red[0];
}

// ---------------------------------------------------------------------------
// Fused norm-scale + temperature + softmax; writes bf16 attn (A of next GEMM)
// ---------------------------------------------------------------------------
__global__ __launch_bounds__(256) void softmax_attn(
    const float* __restrict__ attn_raw, const float* __restrict__ sumsq,
    const float* __restrict__ temp, unsigned short* __restrict__ attn_bf) {
  __shared__ float red[256];
  int row = blockIdx.x, b = row >> 8, c = row & 255;
  int d = threadIdx.x;
  float rq = 1.0f / fmaxf(sqrtf(sumsq[b * 512 + c]), 1e-12f);
  float rk = 1.0f / fmaxf(sqrtf(sumsq[b * 512 + 256 + d]), 1e-12f);
  float v = attn_raw[(long long)row * 256 + d] * rq * rk * temp[0];
  red[d] = v; __syncthreads();
  for (int w = 128; w > 0; w >>= 1) {
    if (d < w) red[d] = fmaxf(red[d], red[d + w]);
    __syncthreads();
  }
  float mx = red[0]; __syncthreads();
  float e = __expf(v - mx);
  red[d] = e; __syncthreads();
  for (int w = 128; w > 0; w >>= 1) {
    if (d < w) red[d] += red[d + w];
    __syncthreads();
  }
  attn_bf[(long long)row * 256 + d] = f2bf(e * (1.0f / red[0]));
}

// ---------------------------------------------------------------------------
extern "C" void kernel_launch(void* const* d_in, const int* in_sizes, int n_in,
                              void* d_out, int out_size, void* d_ws, size_t ws_size,
                              hipStream_t stream) {
  (void)in_sizes; (void)n_in; (void)out_size; (void)ws_size;
  const float* x      = (const float*)d_in[0];
  const float* w_qkv  = (const float*)d_in[1];
  const float* w_dw   = (const float*)d_in[2];
  const float* temp   = (const float*)d_in[3];
  const float* w_proj = (const float*)d_in[4];
  float* out = (float*)d_out;

  const int B = 8, C = 256, N = 16384, M3 = 768;

  char* ws = (char*)d_ws;
  size_t off = 0;
  auto take = [&](size_t bytes) {
    char* p = ws + off;
    off = (off + bytes + 255) & ~(size_t)255;
    return p;
  };
  unsigned short* x_bf     = (unsigned short*)take((size_t)B * C * N * 2);   // 64 MB
  unsigned short* qkv_bf   = (unsigned short*)take((size_t)B * M3 * N * 2);  // 192 MB
  unsigned short* tmp_bf   = (unsigned short*)take((size_t)B * C * N * 2);   // 64 MB
  unsigned short* wqkv_bf  = (unsigned short*)take((size_t)M3 * C * 2);
  unsigned short* wproj_bf = (unsigned short*)take((size_t)C * C * 2);
  float*          sumsq    = (float*)take((size_t)B * 2 * C * 4);
  float*          attn_raw = (float*)take((size_t)B * C * C * 4);            // 2 MB
  unsigned short* attn_bf  = (unsigned short*)take((size_t)B * C * C * 2);

  // 1) bf16 copies of x and the two weight matrices
  long long nx8 = (long long)B * C * N / 8;
  k_f32_to_bf16<<<dim3((unsigned)((nx8 + 255) / 256)), 256, 0, stream>>>(x, x_bf, nx8);
  long long nq8 = (long long)M3 * C / 8;
  k_f32_to_bf16<<<dim3((unsigned)((nq8 + 255) / 256)), 256, 0, stream>>>(w_qkv, wqkv_bf, nq8);
  long long np8 = (long long)C * C / 8;
  k_f32_to_bf16<<<dim3((unsigned)((np8 + 255) / 256)), 256, 0, stream>>>(w_proj, wproj_bf, np8);

  // 2) qkv = (W_qkv @ x) * dw   [M=768, K=256, N=16384], bf16 out
  gemm_rowB<true, true><<<dim3(N / 64, M3 / 128, B), 256, 0, stream>>>(
      wqkv_bf, 0LL, x_bf, (long long)C * N, w_dw,
      qkv_bf, (long long)M3 * N, C, N, N, C);

  // 3) row norms of q,k
  sumsq_rows<<<dim3(B * 2 * C), 256, 0, stream>>>(qkv_bf, sumsq, N);

  // 4) attn_raw = q @ k^T   [256x256 per batch, K=16384]
  attn_qk<<<dim3(C / 64, C / 128, B), 256, 0, stream>>>(qkv_bf, attn_raw, N);

  // 5) fold norms + temperature, softmax, emit bf16
  softmax_attn<<<dim3(B * C), 256, 0, stream>>>(attn_raw, sumsq, temp, attn_bf);

  // 6) tmp = attn @ v   [M=256, K=256, N=16384]
  gemm_rowB<false, true><<<dim3(N / 64, C / 128, B), 256, 0, stream>>>(
      attn_bf, (long long)C * C, qkv_bf + 512LL * N, (long long)M3 * N, nullptr,
      tmp_bf, (long long)C * N, C, N, N, C);

  // 7) out = W_proj @ tmp -> f32 output
  gemm_rowB<false, false><<<dim3(N / 64, C / 128, B), 256, 0, stream>>>(
      wproj_bf, 0LL, tmp_bf, (long long)C * N, nullptr,
      out, (long long)C * N, C, N, N, C);
}
